// GraphGPSLayer_75110388072734
// MI455X (gfx1250) — matrix operand order repaired and checked by reading, not compiled, not run on hardware
//
#include <hip/hip_runtime.h>
#include <math.h>

typedef __attribute__((ext_vector_type(16))) _Float16 v16h;
typedef __attribute__((ext_vector_type(8)))  float    v8f;
typedef unsigned int u32;
typedef unsigned long long u64;

#define N_NODES  16384L
#define DIM      128
#define N_EDGES  524288L
#define NPGC     1024
#define NGRAPH   16

union Frag { v16h h; u32 u[8]; };

__device__ __forceinline__ u32 pack2(float a, float b){
  union { u32 u; _Float16 h[2]; } t;
  t.h[0] = (_Float16)a; t.h[1] = (_Float16)b;
  return t.u;
}
// A-matrix 16x32 f16 k order used here (contiguous, matches B packing): lane lo = row; VGPR v -> K = 16*hi+2v
__device__ __forceinline__ int a_kidx(int v, int hi){
  return hi*16 + 2*v;
}
__device__ __forceinline__ void async_cp16(unsigned lds_byte_off, const void* g){
  asm volatile("global_load_async_to_lds_b128 %0, %1, off"
               :: "v"(lds_byte_off), "v"((u64)g) : "memory");
}
__device__ __forceinline__ unsigned lds_off(const void* p){
  return (unsigned)(u64)p;          // low 32 bits of generic addr = LDS byte offset
}

// ---- Stage a contiguous 16 x (KT*32) f32 strip into padded LDS via async copies
template<int KT>
__device__ __forceinline__ void stage_strip(const float* __restrict__ Xs, float* lds){
  const int l = threadIdx.x & 31;
  const int PITCH = KT*32 + 4;
  const char* g = (const char*)Xs;
  unsigned lbase = lds_off(lds);
  #pragma unroll
  for (int i = 0; i < KT*4; ++i){
    int c = i*32 + l;                       // 16B chunk index
    int r = c / (KT*8), q4 = c % (KT*8);
    async_cp16(lbase + (unsigned)((r*PITCH + q4*4)*4), g + (long)c*16);
  }
  asm volatile("s_wait_asynccnt 0x0" ::: "memory");
}

// ---- 16-row x NT*16 GEMM from LDS-staged A strip, W pre-packed in B-frag order
template<int KT, int NT>
__device__ __forceinline__ void compute_strip(const float* lds, const u32* __restrict__ Wp,
                                              v8f (&acc)[NT]){
  const int l = threadIdx.x & 31, lo = l & 15, hi = l >> 4;
  const int PITCH = KT*32 + 4;
  const v8f vz = {0.f,0.f,0.f,0.f,0.f,0.f,0.f,0.f};
  #pragma unroll
  for (int nt = 0; nt < NT; ++nt) acc[nt] = vz;
  #pragma unroll
  for (int kt = 0; kt < KT; ++kt){
    Frag a;
    const float* xr = lds + lo*PITCH + kt*32;
    #pragma unroll
    for (int v = 0; v < 8; ++v){ int k = a_kidx(v,hi); a.u[v] = pack2(xr[k], xr[k+1]); }
    #pragma unroll
    for (int nt = 0; nt < NT; ++nt){
      Frag b;
      const u32* wb = Wp + ((long)(nt*KT + kt)*32 + l)*8;
      #pragma unroll
      for (int v = 0; v < 8; ++v) b.u[v] = wb[v];
      acc[nt] = __builtin_amdgcn_wmma_f32_16x16x32_f16(false, a.h, false, b.h,
                                                       (short)0, acc[nt], false, false);
    }
  }
}

// ---- Pack W (Dout x K row-major f32) into WMMA B-fragment order (f16 pairs in dwords)
__global__ void k_pack(const float* __restrict__ W, int K, int KT, u32* __restrict__ dst, int total){
  int f = blockIdx.x*256 + threadIdx.x;
  if (f >= total) return;
  int v = f & 7, lane = (f >> 3) & 31, rest = f >> 8;
  int kt = rest % KT, nt = rest / KT;
  int lo = lane & 15, hi = lane >> 4;
  int n  = nt*16 + lo;
  int kk = kt*32 + 16*hi + 2*v;
  dst[f] = pack2(W[(long)n*K + kk], W[(long)n*K + kk + 1]);
}

// ---- Node projections Ax,Bx,Dx,Ex: one block per 16-row strip, loops all 4 matrices
__global__ __launch_bounds__(32) void k_node_proj(const float* __restrict__ x, const u32* __restrict__ wpk,
                            const float* __restrict__ bA, const float* __restrict__ bB,
                            const float* __restrict__ bD, const float* __restrict__ bE,
                            float* __restrict__ outbase)
{
  extern __shared__ float smem[];
  long row0 = (long)blockIdx.x * 16;
  stage_strip<4>(x + row0*DIM, smem);
  const int l = threadIdx.x & 31, lo = l & 15, hi = l >> 4;
  const float* biases[4] = {bA, bB, bD, bE};
  const int widx[4] = {0, 1, 3, 4};              // packed order WA,WB,WC,WD,WE
  #pragma unroll
  for (int mat = 0; mat < 4; ++mat){
    v8f acc[8];
    compute_strip<4,8>(smem, wpk + (long)widx[mat]*8192, acc);
    float* out = outbase + (long)mat * N_NODES * DIM;
    #pragma unroll
    for (int nt = 0; nt < 8; ++nt){
      int c = nt*16 + lo; float bv = biases[mat][c];
      #pragma unroll
      for (int v = 0; v < 8; ++v)
        out[(row0 + v + 8*hi)*DIM + c] = acc[nt][v] + bv;
    }
  }
}

// ---- Fused edge kernel: Ce tile via WMMA, gate, message, scatter-add
__global__ __launch_bounds__(32) void k_edge(const float* __restrict__ ea, const u32* __restrict__ wcp,
                       const float* __restrict__ bC,
                       const int* __restrict__ src, const int* __restrict__ dst,
                       const float* __restrict__ Bx, const float* __restrict__ Dx,
                       const float* __restrict__ Ex,
                       float* __restrict__ agg, float* __restrict__ deg)
{
  extern __shared__ float smem[];
  long e0 = (long)blockIdx.x * 16;
  stage_strip<4>(ea + e0*DIM, smem);
  v8f acc[8];
  compute_strip<4,8>(smem, wcp, acc);
  int l = threadIdx.x & 31, lo = l & 15, hi = l >> 4;
  if (hi == 0) atomicAdd(deg + dst[e0 + lo], 1.0f);
  #pragma unroll
  for (int v = 0; v < 8; ++v){
    long e = e0 + v + 8*hi;
    int sN = src[e], tN = dst[e];
    const float* Dr = Dx + (long)tN*DIM;
    const float* Er = Ex + (long)sN*DIM;
    const float* Br = Bx + (long)sN*DIM;
    float* ar = agg + (long)tN*DIM;
    #pragma unroll
    for (int nt = 0; nt < 8; ++nt){
      int d = nt*16 + lo;
      float ce = acc[nt][v] + bC[d];
      float sg = 1.f / (1.f + __expf(-(ce + Dr[d] + Er[d])));
      atomicAdd(ar + d, sg * Br[d]);
    }
  }
}

__global__ void k_finalize(const float* __restrict__ Ax, const float* __restrict__ agg,
                           const float* __restrict__ deg, float* __restrict__ hpre){
  long i = (long)blockIdx.x*256 + threadIdx.x;
  long row = i >> 7;
  float dv = fmaxf(deg[row], 1.0f);
  hpre[i] = Ax[i] + agg[i] / dv;
}

__global__ void k_bn_partial(const float* __restrict__ hpre, float* __restrict__ part){
  int c = threadIdx.x, chunk = blockIdx.x;
  float s = 0.f, q = 0.f;
  for (int r = 0; r < 128; ++r){
    float v = hpre[((long)chunk*128 + r)*DIM + c];
    s += v; q += v*v;
  }
  part[chunk*128 + c] = s;
  part[16384 + chunk*128 + c] = q;
}

__global__ void k_bn_final(const float* __restrict__ part, const float* __restrict__ g,
                           const float* __restrict__ b, float* __restrict__ scale,
                           float* __restrict__ shift){
  int c = threadIdx.x;
  float s = 0.f, q = 0.f;
  for (int ch = 0; ch < 128; ++ch){ s += part[ch*128 + c]; q += part[16384 + ch*128 + c]; }
  float mu = s * (1.f/16384.f);
  float var = q * (1.f/16384.f) - mu*mu;
  float sc = g[c] * rsqrtf(var + 1e-5f);
  scale[c] = sc; shift[c] = b[c] - mu*sc;
}

__global__ void k_bn_apply(const float* __restrict__ hpre, const float* __restrict__ scale,
                           const float* __restrict__ shift, float* __restrict__ h){
  long i = (long)blockIdx.x*256 + threadIdx.x;
  int c = i & 127;
  h[i] = fmaxf(hpre[i]*scale[c] + shift[c], 0.f);
}

// ---- QKV projection: one block per strip, loops q/k/v segments
__global__ __launch_bounds__(32) void k_qkv(const float* __restrict__ h, const u32* __restrict__ wqkvp,
                      const float* __restrict__ bqkv, float* __restrict__ qkv)
{
  extern __shared__ float smem[];
  long row0 = (long)blockIdx.x * 16;
  stage_strip<4>(h + row0*DIM, smem);
  const int l = threadIdx.x & 31, lo = l & 15, hi = l >> 4;
  #pragma unroll
  for (int seg = 0; seg < 3; ++seg){
    v8f acc[8];
    compute_strip<4,8>(smem, wqkvp + (long)seg*8192, acc);
    #pragma unroll
    for (int nt = 0; nt < 8; ++nt){
      int c = seg*128 + nt*16 + lo; float bv = bqkv[c];
      #pragma unroll
      for (int v = 0; v < 8; ++v)
        qkv[(row0 + v + 8*hi)*384 + c] = acc[nt][v] + bv;
    }
  }
}

// ---- Flash attention: 4 waves/block = 4 query tiles sharing double-buffered K/V in LDS
#define KVP 36
__global__ __launch_bounds__(128) void k_attn(const float* __restrict__ qkv, float* __restrict__ ctx)
{
  const int hh = blockIdx.y, g = blockIdx.z;
  const long base = (long)g * NPGC;
  const int hd = hh * 32;
  const int tid = threadIdx.x;
  const int wave = tid >> 5, l = tid & 31, lo = l & 15, hi = l >> 4;
  const int qt = blockIdx.x*4 + wave;

  __shared__ float Kc[2][32*KVP];
  __shared__ float Vc[2][32*KVP];
  __shared__ u32  Pl[4][256];

  Frag aq;
  const float qscale = 0.17677669529663687f;     // 1/sqrt(32)
  {
    const float* qr = qkv + (base + qt*16 + lo)*384 + hd;
    #pragma unroll
    for (int v = 0; v < 8; ++v){ int k = a_kidx(v,hi); aq.u[v] = pack2(qr[k]*qscale, qr[k+1]*qscale); }
  }
  v8f O0 = {0.f,0.f,0.f,0.f,0.f,0.f,0.f,0.f}, O1 = O0;
  float mrow[8], lrow[8];
  #pragma unroll
  for (int v = 0; v < 8; ++v){ mrow[v] = -3.0e38f; lrow[v] = 0.f; }

  // stage K/V chunk kc into buffer buf (128 threads, 4 async b128 per thread)
  auto stage = [&](int buf, int kc){
    #pragma unroll
    for (int j = 0; j < 2; ++j){
      int c = j*128 + tid;
      int r = c >> 3, q4 = c & 7;
      const float* row = qkv + (base + kc + r)*384 + hd + q4*4;
      async_cp16(lds_off(&Kc[buf][r*KVP + q4*4]), row + 128);
      async_cp16(lds_off(&Vc[buf][r*KVP + q4*4]), row + 256);
    }
  };

  stage(0, 0);
  for (int it = 0; it < 32; ++it){
    int buf = it & 1;
    int kc = it * 32;
    if (it + 1 < 32){
      stage(buf ^ 1, kc + 32);
      asm volatile("s_wait_asynccnt 0x4" ::: "memory");  // current buffer landed, prefetch in flight
    } else {
      asm volatile("s_wait_asynccnt 0x0" ::: "memory");
    }
    __syncthreads();

    Frag bk0, bk1;
    #pragma unroll
    for (int v = 0; v < 8; ++v){
      int kk = 16*hi + 2*v;
      const float* k0 = &Kc[buf][lo*KVP + kk];
      const float* k1 = &Kc[buf][(16 + lo)*KVP + kk];
      bk0.u[v] = pack2(k0[0], k0[1]);
      bk1.u[v] = pack2(k1[0], k1[1]);
    }
    const v8f vz = {0.f,0.f,0.f,0.f,0.f,0.f,0.f,0.f};
    v8f s0 = __builtin_amdgcn_wmma_f32_16x16x32_f16(false, aq.h, false, bk0.h, (short)0, vz, false, false);
    v8f s1 = __builtin_amdgcn_wmma_f32_16x16x32_f16(false, aq.h, false, bk1.h, (short)0, vz, false, false);

    float pm[8];
    #pragma unroll
    for (int v = 0; v < 8; ++v) pm[v] = fmaxf(s0[v], s1[v]);
    #pragma unroll
    for (int m = 1; m < 16; m <<= 1){
      #pragma unroll
      for (int v = 0; v < 8; ++v) pm[v] = fmaxf(pm[v], __shfl_xor(pm[v], m, 32));
    }
    float sc[8], rs[8];
    #pragma unroll
    for (int v = 0; v < 8; ++v){
      float mn = fmaxf(mrow[v], pm[v]);
      sc[v] = __expf(mrow[v] - mn);
      mrow[v] = mn;
      float p0 = __expf(s0[v] - mn), p1 = __expf(s1[v] - mn);
      s0[v] = p0; s1[v] = p1;
      rs[v] = p0 + p1;
    }
    #pragma unroll
    for (int m = 1; m < 16; m <<= 1){
      #pragma unroll
      for (int v = 0; v < 8; ++v) rs[v] += __shfl_xor(rs[v], m, 32);
    }
    #pragma unroll
    for (int v = 0; v < 8; ++v){
      lrow[v] = lrow[v]*sc[v] + rs[v];
      O0[v] *= sc[v]; O1[v] *= sc[v];
    }
    // P (C-layout) -> A-layout, wave-private LDS region, wave-local ordering
    _Float16* Pw = (_Float16*)&Pl[wave][0];
    #pragma unroll
    for (int v = 0; v < 8; ++v){
      int m = v + 8*hi;
      Pw[m*32 + lo]      = (_Float16)s0[v];
      Pw[m*32 + 16 + lo] = (_Float16)s1[v];
    }
    asm volatile("s_wait_dscnt 0x0" ::: "memory");
    Frag pa;
    #pragma unroll
    for (int v = 0; v < 8; ++v){ int k = a_kidx(v,hi); pa.u[v] = Pl[wave][(lo*32 + k) >> 1]; }

    Frag bv0, bv1;
    #pragma unroll
    for (int v = 0; v < 8; ++v){
      int kk = 16*hi + 2*v;
      bv0.u[v] = pack2(Vc[buf][kk*KVP + lo],      Vc[buf][(kk+1)*KVP + lo]);
      bv1.u[v] = pack2(Vc[buf][kk*KVP + 16 + lo], Vc[buf][(kk+1)*KVP + 16 + lo]);
    }
    O0 = __builtin_amdgcn_wmma_f32_16x16x32_f16(false, pa.h, false, bv0.h, (short)0, O0, false, false);
    O1 = __builtin_amdgcn_wmma_f32_16x16x32_f16(false, pa.h, false, bv1.h, (short)0, O1, false, false);

    __syncthreads();    // all waves done with buf before next prefetch overwrites it
  }
  #pragma unroll
  for (int v = 0; v < 8; ++v){
    long node = base + qt*16 + v + 8*hi;
    float inv = 1.f / lrow[v];
    ctx[node*DIM + hd + lo]      = O0[v]*inv;
    ctx[node*DIM + hd + 16 + lo] = O1[v]*inv;
  }
}

// ---- GEMM(128 out) + bias + residual + LayerNorm (Wo/LN1 and FFN2/LN2)
template<int KT>
__global__ __launch_bounds__(32) void k_gemm_ln(const float* __restrict__ X, long ldx,
                        const u32* __restrict__ Wp, const float* __restrict__ bias,
                        const float* __restrict__ res, const float* __restrict__ gma,
                        const float* __restrict__ bta, float* __restrict__ out)
{
  extern __shared__ float smem[];
  long row0 = (long)blockIdx.x * 16;
  stage_strip<KT>(X + row0*ldx, smem);
  v8f acc[8];
  compute_strip<KT,8>(smem, Wp, acc);
  int l = threadIdx.x & 31, lo = l & 15, hi = l >> 4;
  float s[8], q[8], mu[8], inv[8];
  #pragma unroll
  for (int v = 0; v < 8; ++v) s[v] = 0.f;
  #pragma unroll
  for (int nt = 0; nt < 8; ++nt){
    int c = nt*16 + lo; float bv = bias[c];
    #pragma unroll
    for (int v = 0; v < 8; ++v){
      float val = acc[nt][v] + bv + res[(row0 + v + 8*hi)*DIM + c];
      acc[nt][v] = val; s[v] += val;
    }
  }
  #pragma unroll
  for (int m = 1; m < 16; m <<= 1){
    #pragma unroll
    for (int v = 0; v < 8; ++v) s[v] += __shfl_xor(s[v], m, 32);
  }
  #pragma unroll
  for (int v = 0; v < 8; ++v){ mu[v] = s[v]*(1.f/128.f); q[v] = 0.f; }
  #pragma unroll
  for (int nt = 0; nt < 8; ++nt){
    #pragma unroll
    for (int v = 0; v < 8; ++v){ float d = acc[nt][v] - mu[v]; q[v] += d*d; }
  }
  #pragma unroll
  for (int m = 1; m < 16; m <<= 1){
    #pragma unroll
    for (int v = 0; v < 8; ++v) q[v] += __shfl_xor(q[v], m, 32);
  }
  #pragma unroll
  for (int v = 0; v < 8; ++v) inv[v] = rsqrtf(q[v]*(1.f/128.f) + 1e-5f);
  #pragma unroll
  for (int nt = 0; nt < 8; ++nt){
    int c = nt*16 + lo; float gv = gma[c], bv = bta[c];
    #pragma unroll
    for (int v = 0; v < 8; ++v)
      out[(row0 + v + 8*hi)*DIM + c] = (acc[nt][v] - mu[v])*inv[v]*gv + bv;
  }
}

// ---- FFN first GEMM + exact GELU: one block per strip, loops 4 output segments
__global__ __launch_bounds__(32) void k_ffn1(const float* __restrict__ h1, const u32* __restrict__ w1p,
                       const float* __restrict__ b1, float* __restrict__ t)
{
  extern __shared__ float smem[];
  long row0 = (long)blockIdx.x * 16;
  stage_strip<4>(h1 + row0*DIM, smem);
  const int l = threadIdx.x & 31, lo = l & 15, hi = l >> 4;
  #pragma unroll
  for (int seg = 0; seg < 4; ++seg){
    v8f acc[8];
    compute_strip<4,8>(smem, w1p + (long)seg*8192, acc);
    #pragma unroll
    for (int nt = 0; nt < 8; ++nt){
      int c = seg*128 + nt*16 + lo; float bv = b1[c];
      #pragma unroll
      for (int v = 0; v < 8; ++v){
        float xv = acc[nt][v] + bv;
        t[(row0 + v + 8*hi)*512 + c] = 0.5f*xv*(1.f + erff(xv*0.70710678118654752f));
      }
    }
  }
}

extern "C" void kernel_launch(void* const* d_in, const int* in_sizes, int n_in,
                              void* d_out, int out_size, void* d_ws, size_t ws_size,
                              hipStream_t stream) {
  (void)in_sizes; (void)n_in; (void)out_size; (void)ws_size;
  const float* x    = (const float*)d_in[0];
  const float* ea   = (const float*)d_in[1];
  const float* WA   = (const float*)d_in[2];  const float* bA   = (const float*)d_in[3];
  const float* WB   = (const float*)d_in[4];  const float* bB   = (const float*)d_in[5];
  const float* WC   = (const float*)d_in[6];  const float* bC   = (const float*)d_in[7];
  const float* WD   = (const float*)d_in[8];  const float* bD   = (const float*)d_in[9];
  const float* WE   = (const float*)d_in[10]; const float* bE   = (const float*)d_in[11];
  const float* bn_g = (const float*)d_in[12]; const float* bn_b = (const float*)d_in[13];
  const float* Wqkv = (const float*)d_in[14]; const float* bqkv = (const float*)d_in[15];
  const float* Wo   = (const float*)d_in[16]; const float* bo   = (const float*)d_in[17];
  const float* ln1g = (const float*)d_in[18]; const float* ln1b = (const float*)d_in[19];
  const float* W1   = (const float*)d_in[20]; const float* b1   = (const float*)d_in[21];
  const float* W2   = (const float*)d_in[22]; const float* b2   = (const float*)d_in[23];
  const float* ln2g = (const float*)d_in[24]; const float* ln2b = (const float*)d_in[25];
  const int*   eidx = (const int*)d_in[26];
  const int* srcI = eidx;
  const int* dstI = eidx + N_EDGES;
  float* out = (float*)d_out;

  const long ND = N_NODES * DIM;
  char* ws = (char*)d_ws;
  u32* wpk = (u32*)ws;                           // 139264 dwords of packed weights
  float* f = (float*)(ws + 139264*4);
  float* P      = f;                             // Ax,Bx,Dx,Ex contiguous (4*ND)
  float* agg    = P + 4*ND;
  float* deg    = agg + ND;                      // 16384
  float* hpre   = deg + 16384;
  float* h      = hpre + ND;
  float* ctx    = h + ND;
  float* h1     = ctx + ND;
  float* t      = h1 + ND;                       // N*512
  float* bnpart = t + N_NODES*512;               // 32768
  float* bnscale = bnpart + 32768;
  float* bnshift = bnscale + 128;
  float* qkv    = P + ND;                        // reuse Bx..Ex (3*ND = N*384)

  u32* WAp = wpk + 0,     *WBp = wpk + 8192,  *WCp = wpk + 16384;
  u32* WDp = wpk + 24576, *WEp = wpk + 32768;
  u32* Wqkvp = wpk + 40960, *Wop = wpk + 65536;
  u32* W1p = wpk + 73728,  *W2p = wpk + 106496;

  k_pack<<<32, 256, 0, stream>>>(WA, 128, 4, WAp, 8192);
  k_pack<<<32, 256, 0, stream>>>(WB, 128, 4, WBp, 8192);
  k_pack<<<32, 256, 0, stream>>>(WC, 128, 4, WCp, 8192);
  k_pack<<<32, 256, 0, stream>>>(WD, 128, 4, WDp, 8192);
  k_pack<<<32, 256, 0, stream>>>(WE, 128, 4, WEp, 8192);
  k_pack<<<96, 256, 0, stream>>>(Wqkv, 128, 4, Wqkvp, 24576);
  k_pack<<<32, 256, 0, stream>>>(Wo, 128, 4, Wop, 8192);
  k_pack<<<128, 256, 0, stream>>>(W1, 128, 4, W1p, 32768);
  k_pack<<<128, 256, 0, stream>>>(W2, 512, 16, W2p, 32768);

  hipMemsetAsync(agg, 0, (size_t)(ND + 16384) * 4, stream);   // agg + deg

  const size_t LDS4  = 16 * (4*32 + 4) * 4;      // 8448 B
  const size_t LDS16 = 16 * (16*32 + 4) * 4;     // 33024 B

  k_node_proj<<<N_NODES/16, 32, LDS4, stream>>>(x, wpk, bA, bB, bD, bE, P);
  k_edge<<<N_EDGES/16, 32, LDS4, stream>>>(ea, WCp, bC, srcI, dstI,
                                           P + ND, P + 2*ND, P + 3*ND, agg, deg);
  k_finalize<<<ND/256, 256, 0, stream>>>(P, agg, deg, hpre);
  k_bn_partial<<<128, 128, 0, stream>>>(hpre, bnpart);
  k_bn_final<<<1, 128, 0, stream>>>(bnpart, bn_g, bn_b, bnscale, bnshift);
  k_bn_apply<<<ND/256, 256, 0, stream>>>(hpre, bnscale, bnshift, h);
  k_qkv<<<N_NODES/16, 32, LDS4, stream>>>(h, Wqkvp, bqkv, qkv);
  k_attn<<<dim3(NPGC/64, 4, NGRAPH), 128, 0, stream>>>(qkv, ctx);
  k_gemm_ln<4><<<N_NODES/16, 32, LDS4, stream>>>(ctx, DIM, Wop, bo, h, ln1g, ln1b, h1);
  k_ffn1<<<N_NODES/16, 32, LDS4, stream>>>(h1, W1p, b1, t);
  k_gemm_ln<16><<<N_NODES/16, 32, LDS16, stream>>>(t, 512, W2p, b2, h1, ln2g, ln2b, out);
}
